// MultiHeadedAttention_CI_36919538876977
// MI455X (gfx1250) — compile-verified, matrix operand order checked by reading, not run on hardware
//
#include <hip/hip_runtime.h>
#include <hip/hip_bf16.h>

#define BATCH  16
#define S_LEN  1024
#define DMODEL 512
#define HEADS  8
#define DKH    64
#define SW     (S_LEN / 32)   // mask words per row

typedef __attribute__((ext_vector_type(16))) __bf16 v16bf;
typedef __attribute__((ext_vector_type(8)))  float  v8f;

// ---------------------------------------------------------------------------
// 16x32 bf16 A/B fragment layout (CDNA5 ISA 7.12.2): lane element e holds
//   k = (e<8 ? e : e+8) + 8*half      (half = lane>=16)
// so each 8-element group is 16 contiguous bytes at (row, kb + 8*half) and
// (row, kb + 8*half + 16).  `p` must already point at row*ld + kb + 8*half.
// ---------------------------------------------------------------------------
__device__ __forceinline__ v16bf load_frag_bf16(const __bf16* __restrict__ p) {
  union { uint4 u[2]; v16bf v; } x;
  x.u[0] = *(const uint4*)p;          // elements 0..7   -> global_load_b128
  x.u[1] = *(const uint4*)(p + 16);   // elements 8..15  -> global_load_b128
  return x.v;
}

__device__ __forceinline__ v16bf load_frag_f32(const float* __restrict__ p) {
  float4 a0 = *(const float4*)(p);
  float4 a1 = *(const float4*)(p + 4);
  float4 b0 = *(const float4*)(p + 16);
  float4 b1 = *(const float4*)(p + 20);
  v16bf f;
  f[0]  = (__bf16)a0.x; f[1]  = (__bf16)a0.y; f[2]  = (__bf16)a0.z; f[3]  = (__bf16)a0.w;
  f[4]  = (__bf16)a1.x; f[5]  = (__bf16)a1.y; f[6]  = (__bf16)a1.z; f[7]  = (__bf16)a1.w;
  f[8]  = (__bf16)b0.x; f[9]  = (__bf16)b0.y; f[10] = (__bf16)b0.z; f[11] = (__bf16)b0.w;
  f[12] = (__bf16)b1.x; f[13] = (__bf16)b1.y; f[14] = (__bf16)b1.z; f[15] = (__bf16)b1.w;
  return f;
}

__device__ __forceinline__ v8f wmma_bf16(v16bf a, v16bf b, v8f c) {
  // D = A(16x32) * B(32x16) + C, f32 accumulate
  return __builtin_amdgcn_wmma_f32_16x16x32_bf16(false, a, false, b,
                                                 (short)0, c, false, false);
}

// ---------------------------------------------------------------------------
// Kernel 0: pack int mask [B,S,S] -> bit mask [B,S,S/32]
// ---------------------------------------------------------------------------
__global__ void maskpack_kernel(const int* __restrict__ mask,
                                unsigned* __restrict__ bits, int nwords) {
  int t = blockIdx.x * blockDim.x + threadIdx.x;
  if (t >= nwords) return;
  const int4* p = (const int4*)(mask + (size_t)t * 32);
  unsigned w = 0;
#pragma unroll
  for (int c = 0; c < 8; ++c) {
    int4 v = p[c];
    w |= (unsigned)(v.x != 0) << (c * 4 + 0);
    w |= (unsigned)(v.y != 0) << (c * 4 + 1);
    w |= (unsigned)(v.z != 0) << (c * 4 + 2);
    w |= (unsigned)(v.w != 0) << (c * 4 + 3);
  }
  bits[t] = w;
}

// ---------------------------------------------------------------------------
// Kernel 1: projection  out = x @ W.T + b   (x, W fp32 -> bf16 WMMA)
// 32x64 output tile per wave: two A fragments reuse each W fragment
// (8 WMMAs per 32-wide k-chunk), halving weight traffic vs a 16-row tile.
//   transposed==0: out bf16 [B,H,S,DK]   (Q, K)
//   transposed==1: out bf16 [B,H,DK,S]   (V, pre-transposed for PV WMMA)
// ---------------------------------------------------------------------------
__global__ __launch_bounds__(32) void proj_kernel(
    const float* __restrict__ X, const float* __restrict__ W,
    const float* __restrict__ bias, __bf16* __restrict__ out, int transposed) {
  const int lane = threadIdx.x & 31;
  const int half = lane >> 4;
  const int r    = lane & 15;
  const int r0 = blockIdx.x << 5;   // 32-row tile over B*S rows
  const int nb = blockIdx.y;        // 64-col block == head index
  const int h8 = half << 3;

  v8f zero = {};
  v8f acc[2][4];
#pragma unroll
  for (int mt = 0; mt < 2; ++mt)
#pragma unroll
    for (int t = 0; t < 4; ++t) acc[mt][t] = zero;

#pragma unroll 1
  for (int kc = 0; kc < DMODEL / 32; ++kc) {
    const int kb = kc * 32 + h8;
    v16bf a0 = load_frag_f32(X + (size_t)(r0 + r) * DMODEL + kb);
    v16bf a1 = load_frag_f32(X + (size_t)(r0 + 16 + r) * DMODEL + kb);
#pragma unroll
    for (int t = 0; t < 4; ++t) {
      v16bf bf = load_frag_f32(W + (size_t)(nb * 64 + t * 16 + r) * DMODEL + kb);
      acc[0][t] = wmma_bf16(a0, bf, acc[0][t]);
      acc[1][t] = wmma_bf16(a1, bf, acc[1][t]);
    }
  }

  const int b  = r0 / S_LEN;
  const int s0 = r0 % S_LEN;
#pragma unroll
  for (int mt = 0; mt < 2; ++mt)
#pragma unroll
    for (int t = 0; t < 4; ++t) {
      float bj = bias[nb * 64 + t * 16 + r];
#pragma unroll
      for (int i = 0; i < 8; ++i) {
        int m = mt * 16 + i + h8;     // C layout: VGPR i -> row i + 8*half
        float v = acc[mt][t][i] + bj;
        size_t addr;
        if (transposed)
          addr = (((size_t)b * HEADS + nb) * DKH + (t * 16 + r)) * S_LEN + (s0 + m);
        else
          addr = (((size_t)b * HEADS + nb) * S_LEN + (s0 + m)) * DKH + (t * 16 + r);
        out[addr] = (__bf16)v;
      }
    }
}

// ---------------------------------------------------------------------------
// Kernel 2: dual-stream flash attention.
// One wave per block, one (b, h, 16-row q-tile). 32 keys per iteration.
// pass 0: kv batch = b (homogeneous); pass 1: kv batch = b^1 (heterogeneous).
// combine = ctx0 + 0.5 * relu(ctx1), written bf16 to [B,S,D].
// ---------------------------------------------------------------------------
__global__ __launch_bounds__(32) void attn_kernel(
    const __bf16* __restrict__ Q,      // [B,H,S,DK]
    const __bf16* __restrict__ K,      // [B,H,S,DK]
    const __bf16* __restrict__ VT,     // [B,H,DK,S]
    const unsigned* __restrict__ mbits,// [B,S,S/32]
    __bf16* __restrict__ combine) {    // [B,S,D]
  __shared__ __bf16 lds_p[16][32];     // P tile transpose buffer (C -> A layout)

  const int lane = threadIdx.x & 31;
  const int half = lane >> 4;
  const int r    = lane & 15;
  const int h8   = half << 3;
  const int qt = blockIdx.x & 63;
  const int h  = (blockIdx.x >> 6) & 7;
  const int b  = blockIdx.x >> 9;
  const int q0 = qt << 4;

  const __bf16* qh = Q + ((size_t)b * HEADS + h) * S_LEN * DKH;
  v16bf qa0 = load_frag_bf16(qh + (size_t)(q0 + r) * DKH + h8);
  v16bf qa1 = load_frag_bf16(qh + (size_t)(q0 + r) * DKH + 32 + h8);

  const unsigned* mrow = mbits + ((size_t)b * S_LEN + q0) * SW;

  v8f zero = {};
  v8f ctx0[4];
  v8f acc[4];
  float mi[8], li[8];

#pragma unroll 1
  for (int pass = 0; pass < 2; ++pass) {
    const int kvb = b ^ pass;
    const __bf16* kh = K  + ((size_t)kvb * HEADS + h) * S_LEN * DKH;
    const __bf16* vh = VT + ((size_t)kvb * HEADS + h) * DKH * S_LEN;
#pragma unroll
    for (int t = 0; t < 4; ++t) acc[t] = zero;
#pragma unroll
    for (int i = 0; i < 8; ++i) { mi[i] = -__builtin_inff(); li[i] = 0.f; }

#pragma unroll 1
    for (int kt = 0; kt < S_LEN / 32; ++kt) {
      const int kbase = kt * 32;
      if (kt + 1 < S_LEN / 32) {      // prefetch next K tile -> global_prefetch_b8
        __builtin_prefetch(kh + (size_t)(kbase + 32 + lane) * DKH, 0, 3);
      }
      // ---- scores: two 16x16 tiles over 32 keys, K-dim = 64 ----
      v8f s0 = zero, s1 = zero;
      {
        const __bf16* krow0 = kh + (size_t)(kbase + r) * DKH + h8;
        const __bf16* krow1 = kh + (size_t)(kbase + 16 + r) * DKH + h8;
        s0 = wmma_bf16(qa0, load_frag_bf16(krow0), s0);
        s0 = wmma_bf16(qa1, load_frag_bf16(krow0 + 32), s0);
        s1 = wmma_bf16(qa0, load_frag_bf16(krow1), s1);
        s1 = wmma_bf16(qa1, load_frag_bf16(krow1 + 32), s1);
      }
      // ---- online softmax (row stats replicated across each 16-lane half) --
      float corr[8];
#pragma unroll
      for (int i = 0; i < 8; ++i) {
        const int m = i + h8;
        unsigned w = mrow[(size_t)m * SW + kt];
        float s0v = (float)s0[i] * 0.125f;   // 1/sqrt(64)
        float s1v = (float)s1[i] * 0.125f;
        if (!((w >> r) & 1u))        s0v = -1e9f;
        if (!((w >> (r + 16)) & 1u)) s1v = -1e9f;
        float mloc = fmaxf(s0v, s1v);
#pragma unroll
        for (int off = 1; off < 16; off <<= 1)
          mloc = fmaxf(mloc, __shfl_xor(mloc, off, 32));
        float mnew = fmaxf(mi[i], mloc);
        float c  = __expf(mi[i] - mnew);
        float p0 = __expf(s0v - mnew);
        float p1 = __expf(s1v - mnew);
        float rs = p0 + p1;
#pragma unroll
        for (int off = 1; off < 16; off <<= 1)
          rs += __shfl_xor(rs, off, 32);
        li[i] = li[i] * c + rs;
        mi[i] = mnew;
        corr[i] = c;
        lds_p[m][r]      = (__bf16)p0;
        lds_p[m][r + 16] = (__bf16)p1;
      }
#pragma unroll
      for (int t = 0; t < 4; ++t)
#pragma unroll
        for (int i = 0; i < 8; ++i) acc[t][i] *= corr[i];
      __syncthreads();               // single-wave group: cheap + ds ordering
      // ---- P (C layout) -> A layout: two ds_load_b128 per lane ----
      v16bf pa;
      {
        union { uint4 u[2]; v16bf v; } px;
        px.u[0] = *(const uint4*)&lds_p[r][h8];
        px.u[1] = *(const uint4*)&lds_p[r][h8 + 16];
        pa = px.v;
      }
      // ---- context += P(16x32) * V(32x64); V pre-transposed in memory ----
#pragma unroll
      for (int t = 0; t < 4; ++t) {
        v16bf vb = load_frag_bf16(vh + (size_t)(t * 16 + r) * S_LEN + kbase + h8);
        acc[t] = wmma_bf16(pa, vb, acc[t]);
      }
      __syncthreads();
    }
    // normalize by row sums
#pragma unroll
    for (int t = 0; t < 4; ++t)
#pragma unroll
      for (int i = 0; i < 8; ++i) acc[t][i] = acc[t][i] / li[i];
    if (pass == 0) {
#pragma unroll
      for (int t = 0; t < 4; ++t) ctx0[t] = acc[t];
    }
  }

  // combine = homo + 0.5*relu(heter), store bf16 row-major [B,S,D]
  __bf16* orow = combine + ((size_t)b * S_LEN + q0) * DMODEL + h * DKH;
#pragma unroll
  for (int t = 0; t < 4; ++t)
#pragma unroll
    for (int i = 0; i < 8; ++i) {
      int m = i + h8;
      float v = ctx0[t][i] + 0.5f * fmaxf(acc[t][i], 0.f);
      orow[(size_t)m * DMODEL + t * 16 + r] = (__bf16)v;
    }
}

// ---------------------------------------------------------------------------
// Kernel 3: output projection  out = combine @ Wo.T + bo   (fp32 out)
// 32x64 tile per wave, bf16 A-side loads are pure b128 (no cvt).
// ---------------------------------------------------------------------------
__global__ __launch_bounds__(32) void oproj_kernel(
    const __bf16* __restrict__ X, const float* __restrict__ W,
    const float* __restrict__ bias, float* __restrict__ out) {
  const int lane = threadIdx.x & 31;
  const int half = lane >> 4;
  const int r    = lane & 15;
  const int h8   = half << 3;
  const int r0 = blockIdx.x << 5;
  const int nb = blockIdx.y;

  v8f zero = {};
  v8f acc[2][4];
#pragma unroll
  for (int mt = 0; mt < 2; ++mt)
#pragma unroll
    for (int t = 0; t < 4; ++t) acc[mt][t] = zero;

#pragma unroll 1
  for (int kc = 0; kc < DMODEL / 32; ++kc) {
    const int kb = kc * 32 + h8;
    v16bf a0 = load_frag_bf16(X + (size_t)(r0 + r) * DMODEL + kb);
    v16bf a1 = load_frag_bf16(X + (size_t)(r0 + 16 + r) * DMODEL + kb);
#pragma unroll
    for (int t = 0; t < 4; ++t) {
      v16bf bf = load_frag_f32(W + (size_t)(nb * 64 + t * 16 + r) * DMODEL + kb);
      acc[0][t] = wmma_bf16(a0, bf, acc[0][t]);
      acc[1][t] = wmma_bf16(a1, bf, acc[1][t]);
    }
  }
#pragma unroll
  for (int mt = 0; mt < 2; ++mt)
#pragma unroll
    for (int t = 0; t < 4; ++t) {
      float bj = bias[nb * 64 + t * 16 + r];
#pragma unroll
      for (int i = 0; i < 8; ++i) {
        int m = mt * 16 + i + h8;
        out[(size_t)(r0 + m) * DMODEL + nb * 64 + t * 16 + r] = acc[mt][t][i] + bj;
      }
    }
}

// ---------------------------------------------------------------------------
extern "C" void kernel_launch(void* const* d_in, const int* in_sizes, int n_in,
                              void* d_out, int out_size, void* d_ws,
                              size_t ws_size, hipStream_t stream) {
  const float* query = (const float*)d_in[0];
  const float* key   = (const float*)d_in[1];
  const float* value = (const float*)d_in[2];
  const int*   mask  = (const int*)d_in[3];
  const float* Wq = (const float*)d_in[4];  const float* bq = (const float*)d_in[5];
  const float* Wk = (const float*)d_in[6];  const float* bk = (const float*)d_in[7];
  const float* Wv = (const float*)d_in[8];  const float* bv = (const float*)d_in[9];
  const float* Wo = (const float*)d_in[10]; const float* bo = (const float*)d_in[11];
  float* out = (float*)d_out;

  char* ws = (char*)d_ws;
  const size_t nqkv = (size_t)BATCH * HEADS * S_LEN * DKH;   // 8,388,608
  __bf16* qb   = (__bf16*)ws;  ws += nqkv * sizeof(__bf16);
  __bf16* kb   = (__bf16*)ws;  ws += nqkv * sizeof(__bf16);
  __bf16* vT   = (__bf16*)ws;  ws += nqkv * sizeof(__bf16);
  __bf16* comb = (__bf16*)ws;  ws += (size_t)BATCH * S_LEN * DMODEL * sizeof(__bf16);
  unsigned* mbits = (unsigned*)ws;

  const int nwords = BATCH * S_LEN * SW;
  maskpack_kernel<<<(nwords + 255) / 256, 256, 0, stream>>>(mask, mbits, nwords);

  dim3 pgrid(BATCH * S_LEN / 32, DMODEL / 64);
  proj_kernel<<<pgrid, 32, 0, stream>>>(query, Wq, bq, qb, 0);
  proj_kernel<<<pgrid, 32, 0, stream>>>(key,   Wk, bk, kb, 0);
  proj_kernel<<<pgrid, 32, 0, stream>>>(value, Wv, bv, vT, 1);

  attn_kernel<<<BATCH * HEADS * (S_LEN / 16), 32, 0, stream>>>(qb, kb, vT,
                                                               mbits, comb);

  oproj_kernel<<<pgrid, 32, 0, stream>>>(comb, Wo, bo, out);
}
